// STPTransformer_38706245272402
// MI455X (gfx1250) — matrix operand order repaired and checked by reading, not compiled, not run on hardware
//
#include <hip/hip_runtime.h>

typedef __attribute__((ext_vector_type(16))) _Float16 v16h;
typedef __attribute__((ext_vector_type(8)))  float    v8f;

union V16H { v16h v; _Float16 e[16]; };
union V8F  { v8f  v; float    e[8];  };

#define T_DIM 128
#define D_DIM 64
#define DFF_DIM 256
#define NTHREADS 256

// ---- LDS pool layout (bytes), total 157696 (< 160KB gfx1250 addressable) ----
#define OFF_X    0               // f32 [128][64]  x / x+y / x1 / x1+f   32768
#define OFF_RBIG 32768           // 65536 multi-use
#define OFF_Q    (OFF_RBIG + 0)      // f16 [128][64]  Q   (later sY)
#define OFF_K    (OFF_RBIG + 16384)  // f16 [128][64]  K   (later sY2)
#define OFF_VT   (OFF_RBIG + 32768)  // f16 [64][128]  V^T
#define OFF_M    (OFF_RBIG + 49152)  // f16 [128][64]  Shaw M (j padded to 64)
#define OFF_H    (OFF_RBIG + 0)      // f16 [128][256] FFN hidden (aliases above)
#define OFF_W    98304           // 32768: weight staging <-> sA (f16 [128][128])
#define OFF_AP   98304           // probs share region with weight staging
#define OFF_P    131072          // f16 [128][48]  P = Q @ rel_k^T        12288
#define OFF_RK   143360          // f16 [48][64]   rel_k (rows j, padded)  6144
#define OFF_RVT  149504          // f16 [64][64]   rel_v^T [d][j] (padded) 8192
#define LDS_TOTAL 157696

__device__ __forceinline__ v8f wmma_f16(v16h a, v16h b, v8f c) {
  return __builtin_amdgcn_wmma_f32_16x16x32_f16(false, a, false, b, (short)0, c,
                                                false, false);
}

// A fragment from row-major f16: lane holds row lane&15; lane-half selects
// K sub-blocks {kh*8..+7} and {16+kh*8..+7} (ISA 16-bit A 16x32 layout).
__device__ __forceinline__ v16h frag_a_f16(const _Float16* A, int lda, int m0,
                                           int k0, int lane) {
  const int r  = lane & 15;
  const int kh = (lane >> 4) << 3;
  const _Float16* p = A + (m0 + r) * lda + k0 + kh;
  V16H a;
#pragma unroll
  for (int i = 0; i < 8; ++i) { a.e[i] = p[i]; a.e[8 + i] = p[16 + i]; }
  return a.v;
}

__device__ __forceinline__ v16h frag_a_f32(const float* A, int lda, int m0,
                                           int k0, int lane) {
  const int r  = lane & 15;
  const int kh = (lane >> 4) << 3;
  const float* p = A + (m0 + r) * lda + k0 + kh;
  V16H a;
#pragma unroll
  for (int i = 0; i < 8; ++i) {
    a.e[i]     = (_Float16)p[i];
    a.e[8 + i] = (_Float16)p[16 + i];
  }
  return a.v;
}

// B fragment from TRANSPOSED (N-major [N][K]) f16: lane holds column lane&15;
// lane-half selects K sub-blocks {kh*8..+7} and {16+kh*8..+7} (matches A).
__device__ __forceinline__ v16h frag_b_t(const _Float16* Bt, int ldb, int n0,
                                         int k0, int lane) {
  const int n  = lane & 15;
  const int kh = (lane >> 4) << 3;
  const _Float16* p = Bt + (n0 + n) * ldb + k0 + kh;
  V16H b;
#pragma unroll
  for (int i = 0; i < 8; ++i) { b.e[i] = p[i]; b.e[8 + i] = p[16 + i]; }
  return b.v;
}

// Stage W [KI][NO] (row-major f32 global) as W^T [NO][KI] f16 in LDS.
__device__ __forceinline__ void stage_wT(const float* __restrict__ W,
                                         _Float16* Wt, int KI, int NO, int tid) {
  for (int i = tid; i < KI * NO; i += NTHREADS) {
    int n = i / KI, k = i - n * KI;
    Wt[i] = (_Float16)W[k * NO + n];
  }
}

// 128x64 GEMM (K=64): C = A32 @ Bt + bias. MODE 0: row-major f16 store (ld 64).
// MODE 1: transposed f16 store (ld 128). Compile-time MODE => branch-free stores.
template <int MODE>
__device__ __forceinline__ void gemm_xw64(const float* A32, const _Float16* Bt,
                                          const float* __restrict__ bias,
                                          _Float16* dst, int m0, int cN, int cH,
                                          int lane) {
#pragma unroll
  for (int nt = 0; nt < 4; ++nt) {
    V8F c;
#pragma unroll
    for (int i = 0; i < 8; ++i) c.e[i] = 0.f;
#pragma unroll
    for (int kk = 0; kk < 2; ++kk)
      c.v = wmma_f16(frag_a_f32(A32, 64, m0, kk * 32, lane),
                     frag_b_t(Bt, 64, nt * 16, kk * 32, lane), c.v);
    const float bb = bias[nt * 16 + cN];
#pragma unroll
    for (int r = 0; r < 8; ++r) {
      const int row = m0 + r + cH, col = nt * 16 + cN;
      const _Float16 val = (_Float16)(c.e[r] + bb);
      if (MODE == 0) dst[row * 64 + col] = val;
      else           dst[col * 128 + row] = val;
    }
  }
}

// 128x64 GEMM (K=64) with f16 A, row-major f16 output.
__device__ __forceinline__ void gemm_ff64(const _Float16* A, const _Float16* Bt,
                                          const float* __restrict__ bias,
                                          _Float16* dst, int m0, int cN, int cH,
                                          int lane) {
#pragma unroll
  for (int nt = 0; nt < 4; ++nt) {
    V8F c;
#pragma unroll
    for (int i = 0; i < 8; ++i) c.e[i] = 0.f;
#pragma unroll
    for (int kk = 0; kk < 2; ++kk)
      c.v = wmma_f16(frag_a_f16(A, 64, m0, kk * 32, lane),
                     frag_b_t(Bt, 64, nt * 16, kk * 32, lane), c.v);
    const float bb = bias[nt * 16 + cN];
#pragma unroll
    for (int r = 0; r < 8; ++r)
      dst[(m0 + r + cH) * 64 + nt * 16 + cN] = (_Float16)(c.e[r] + bb);
  }
}

__global__ __launch_bounds__(NTHREADS) void stp_layer_kernel(
    const float* __restrict__ xg,
    const float* __restrict__ Wq,  const float* __restrict__ bq,
    const float* __restrict__ Wk,  const float* __restrict__ bk,
    const float* __restrict__ Wv,  const float* __restrict__ bv,
    const float* __restrict__ Woa, const float* __restrict__ boa,
    const float* __restrict__ Wo,  const float* __restrict__ bo,
    const float* __restrict__ relk, const float* __restrict__ relv,
    const float* __restrict__ ln1g, const float* __restrict__ ln1b,
    const float* __restrict__ ln2g, const float* __restrict__ ln2b,
    const float* __restrict__ Wf1, const float* __restrict__ bf1,
    const float* __restrict__ Wf2, const float* __restrict__ bf2,
    float* __restrict__ outg) {
  __shared__ __align__(16) char smem[LDS_TOTAL];
  float*    sX   = (float*)(smem + OFF_X);
  _Float16* sQ   = (_Float16*)(smem + OFF_Q);
  _Float16* sK   = (_Float16*)(smem + OFF_K);
  _Float16* sVt  = (_Float16*)(smem + OFF_VT);
  _Float16* sM   = (_Float16*)(smem + OFF_M);
  _Float16* sH   = (_Float16*)(smem + OFF_H);
  _Float16* sW   = (_Float16*)(smem + OFF_W);
  _Float16* sA   = (_Float16*)(smem + OFF_AP);
  _Float16* sP   = (_Float16*)(smem + OFF_P);
  _Float16* sRk  = (_Float16*)(smem + OFF_RK);
  _Float16* sRvT = (_Float16*)(smem + OFF_RVT);
  _Float16* sY   = sQ;   // attn output, reuses Q region
  _Float16* sY2  = sK;   // fc_o output, reuses K region

  const int tid  = threadIdx.x;
  const int lane = tid & 31;
  const int wv   = tid >> 5;          // wave id 0..7
  const int m0   = wv << 4;           // this wave's 16 output rows
  const int cN   = lane & 15;         // C-fragment column within tile
  const int cH   = (lane >> 4) << 3;  // C-fragment row offset (0 or 8)
  const float* x = xg + (size_t)blockIdx.x * (T_DIM * D_DIM);
  float* out     = outg + (size_t)blockIdx.x * (T_DIM * D_DIM);

  // ---- prefetch the big (L2-resident) FFN weights: global_prefetch_b8 ----
  for (int i = tid; i < 512; i += NTHREADS) {
    __builtin_prefetch(((const char*)Wf1) + (size_t)i * 128, 0, 1);
    __builtin_prefetch(((const char*)Wf2) + (size_t)i * 128, 0, 1);
  }

  // ---- load x tile (vectorized b128) + rel tables (WMMA-padded) ----
  {
    const float4* xv = (const float4*)x;
    float4* sXv = (float4*)sX;
    for (int i = tid; i < (T_DIM * D_DIM) / 4; i += NTHREADS) sXv[i] = xv[i];
  }
  for (int i = tid; i < 48 * 64; i += NTHREADS) {  // sRk[j][d], j padded to 48
    int j = i >> 6, d = i & 63;
    sRk[i] = (_Float16)((j < 33) ? relk[j * 64 + d] : 0.f);
  }
  for (int i = tid; i < 64 * 64; i += NTHREADS) {  // sRvT[d][j], j padded to 64
    int d = i >> 6, j = i & 63;
    sRvT[i] = (_Float16)((j < 33) ? relv[j * 64 + d] : 0.f);
  }
  __syncthreads();

  // ---- Q = x@Wq+bq, K = x@Wk+bk (row-major f16), V^T = (x@Wv+bv)^T ----
  stage_wT(Wq, sW, 64, 64, tid);
  __syncthreads();
  gemm_xw64<0>(sX, sW, bq, sQ, m0, cN, cH, lane);
  __syncthreads();
  stage_wT(Wk, sW, 64, 64, tid);
  __syncthreads();
  gemm_xw64<0>(sX, sW, bk, sK, m0, cN, cH, lane);
  __syncthreads();
  stage_wT(Wv, sW, 64, 64, tid);
  __syncthreads();
  gemm_xw64<1>(sX, sW, bv, sVt, m0, cN, cH, lane);  // transposed store
  __syncthreads();

  // ---- P = Q @ rel_k^T  ([128][48] f16) ----
#pragma unroll
  for (int nt = 0; nt < 3; ++nt) {
    V8F c;
#pragma unroll
    for (int i = 0; i < 8; ++i) c.e[i] = 0.f;
#pragma unroll
    for (int kk = 0; kk < 2; ++kk)
      c.v = wmma_f16(frag_a_f16(sQ, 64, m0, kk * 32, lane),
                     frag_b_t(sRk, 64, nt * 16, kk * 32, lane), c.v);
#pragma unroll
    for (int r = 0; r < 8; ++r)
      sP[(m0 + r + cH) * 48 + nt * 16 + cN] = (_Float16)c.e[r];
  }
  __syncthreads();

  // ---- logits in registers: Q K^T + gather(P), scale, softmax ----
  V8F acc[8];
#pragma unroll
  for (int j = 0; j < 8; ++j) {
#pragma unroll
    for (int i = 0; i < 8; ++i) acc[j].e[i] = 0.f;
#pragma unroll
    for (int kk = 0; kk < 2; ++kk)
      acc[j].v = wmma_f16(frag_a_f16(sQ, 64, m0, kk * 32, lane),
                          frag_b_t(sK, 64, j * 16, kk * 32, lane), acc[j].v);
  }
#pragma unroll
  for (int j = 0; j < 8; ++j)
#pragma unroll
    for (int r = 0; r < 8; ++r) {
      const int t = m0 + r + cH, u = j * 16 + cN;
      int d = u - t; d = (d < -16) ? -16 : (d > 16) ? 16 : d;
      acc[j].e[r] = (acc[j].e[r] + (float)sP[t * 48 + d + 16]) * 0.125f;
    }
#pragma unroll
  for (int r = 0; r < 8; ++r) {
    float mx = -1e30f;
#pragma unroll
    for (int j = 0; j < 8; ++j) mx = fmaxf(mx, acc[j].e[r]);
#pragma unroll
    for (int msk = 1; msk < 16; msk <<= 1) mx = fmaxf(mx, __shfl_xor(mx, msk, 32));
    float sm = 0.f;
#pragma unroll
    for (int j = 0; j < 8; ++j) {
      const float e = __expf(acc[j].e[r] - mx);
      acc[j].e[r] = e; sm += e;
    }
#pragma unroll
    for (int msk = 1; msk < 16; msk <<= 1) sm += __shfl_xor(sm, msk, 32);
    const float inv = 1.f / sm;
    const int t = m0 + r + cH;
#pragma unroll
    for (int j = 0; j < 8; ++j)
      sA[t * 128 + j * 16 + cN] = (_Float16)(acc[j].e[r] * inv);
  }
  __syncthreads();

  // ---- build Shaw M [128][64]: interior diagonal gather + edge prefix sums
  for (int i = lane; i < 16 * 64; i += 32) {
    const int t = m0 + (i >> 6), j = i & 63;
    float val = 0.f;
    if (j >= 1 && j <= 31) {
      const int u = t + j - 16;
      if (u >= 0 && u < 128) val = (float)sA[t * 128 + u];
    } else if (j == 0) {
      for (int u = 0; u <= t - 16; ++u) val += (float)sA[t * 128 + u];
    } else if (j == 32) {
      for (int u = t + 16; u < 128; ++u) val += (float)sA[t * 128 + u];
    }
    sM[t * 64 + j] = (_Float16)val;
  }
  __syncthreads();

  // ---- y = attn@V + M@rel_v -> sY ----
#pragma unroll
  for (int nt = 0; nt < 4; ++nt) {
    V8F c;
#pragma unroll
    for (int i = 0; i < 8; ++i) c.e[i] = 0.f;
#pragma unroll
    for (int kk = 0; kk < 4; ++kk)            // K = 128 over attn
      c.v = wmma_f16(frag_a_f16(sA, 128, m0, kk * 32, lane),
                     frag_b_t(sVt, 128, nt * 16, kk * 32, lane), c.v);
#pragma unroll
    for (int kk = 0; kk < 2; ++kk)            // K = 64 (33 padded) over M
      c.v = wmma_f16(frag_a_f16(sM, 64, m0, kk * 32, lane),
                     frag_b_t(sRvT, 64, nt * 16, kk * 32, lane), c.v);
#pragma unroll
    for (int r = 0; r < 8; ++r)
      sY[(m0 + r + cH) * 64 + nt * 16 + cN] = (_Float16)c.e[r];
  }
  __syncthreads();

  // ---- fc_o: t1 = y@Woa + boa -> sY2 ----
  stage_wT(Woa, sW, 64, 64, tid);   // sA dead -> reuse region
  __syncthreads();
  gemm_ff64(sY, sW, boa, sY2, m0, cN, cH, lane);
  __syncthreads();

  // ---- outer proj: sX += t1@Wo + bo (residual, in place) ----
  stage_wT(Wo, sW, 64, 64, tid);
  __syncthreads();
#pragma unroll
  for (int nt = 0; nt < 4; ++nt) {
    V8F c;
#pragma unroll
    for (int i = 0; i < 8; ++i) c.e[i] = 0.f;
#pragma unroll
    for (int kk = 0; kk < 2; ++kk)
      c.v = wmma_f16(frag_a_f16(sY2, 64, m0, kk * 32, lane),
                     frag_b_t(sW, 64, nt * 16, kk * 32, lane), c.v);
    const float bb = bo[nt * 16 + cN];
#pragma unroll
    for (int r = 0; r < 8; ++r) {
      const int idx = (m0 + r + cH) * 64 + nt * 16 + cN;
      sX[idx] = sX[idx] + c.e[r] + bb;
    }
  }
  __syncthreads();

  // ---- LayerNorm 1 (x1 overwrites sX) ----
  if (tid < T_DIM) {
    float mu = 0.f;
    for (int d = 0; d < 64; ++d) mu += sX[tid * 64 + d];
    mu *= (1.f / 64.f);
    float var = 0.f;
    for (int d = 0; d < 64; ++d) {
      const float dv = sX[tid * 64 + d] - mu;
      var += dv * dv;
    }
    const float rs = rsqrtf(var * (1.f / 64.f) + 1e-5f);
    for (int d = 0; d < 64; ++d)
      sX[tid * 64 + d] = (sX[tid * 64 + d] - mu) * rs * ln1g[d] + ln1b[d];
  }
  __syncthreads();

  // ---- FFN1: h = relu(x1@Wf1 + bf1) -> sH [128][256] f16 ----
  stage_wT(Wf1, sW, 64, DFF_DIM, tid);   // Wf1^T [256][64]
  __syncthreads();
#pragma unroll
  for (int nt = 0; nt < 16; ++nt) {
    V8F c;
#pragma unroll
    for (int i = 0; i < 8; ++i) c.e[i] = 0.f;
#pragma unroll
    for (int kk = 0; kk < 2; ++kk)
      c.v = wmma_f16(frag_a_f32(sX, 64, m0, kk * 32, lane),
                     frag_b_t(sW, 64, nt * 16, kk * 32, lane), c.v);
    const float bb = bf1[nt * 16 + cN];
#pragma unroll
    for (int r = 0; r < 8; ++r)
      sH[(m0 + r + cH) * DFF_DIM + nt * 16 + cN] =
          (_Float16)fmaxf(c.e[r] + bb, 0.f);
  }
  __syncthreads();

  // ---- FFN2: sX += h@Wf2 + bf2 (residual, in place) ----
  stage_wT(Wf2, sW, DFF_DIM, 64, tid);   // Wf2^T [64][256]
  __syncthreads();
#pragma unroll
  for (int nt = 0; nt < 4; ++nt) {
    V8F c;
#pragma unroll
    for (int i = 0; i < 8; ++i) c.e[i] = 0.f;
#pragma unroll
    for (int kk = 0; kk < 8; ++kk)       // K = 256
      c.v = wmma_f16(frag_a_f16(sH, DFF_DIM, m0, kk * 32, lane),
                     frag_b_t(sW, DFF_DIM, nt * 16, kk * 32, lane), c.v);
    const float bb = bf2[nt * 16 + cN];
#pragma unroll
    for (int r = 0; r < 8; ++r) {
      const int idx = (m0 + r + cH) * 64 + nt * 16 + cN;
      sX[idx] = sX[idx] + c.e[r] + bb;
    }
  }
  __syncthreads();

  // ---- LayerNorm 2 -> global out ----
  if (tid < T_DIM) {
    float mu = 0.f;
    for (int d = 0; d < 64; ++d) mu += sX[tid * 64 + d];
    mu *= (1.f / 64.f);
    float var = 0.f;
    for (int d = 0; d < 64; ++d) {
      const float dv = sX[tid * 64 + d] - mu;
      var += dv * dv;
    }
    const float rs = rsqrtf(var * (1.f / 64.f) + 1e-5f);
    for (int d = 0; d < 64; ++d)
      out[tid * 64 + d] = (sX[tid * 64 + d] - mu) * rs * ln2g[d] + ln2b[d];
  }
}

extern "C" void kernel_launch(void* const* d_in, const int* in_sizes, int n_in,
                              void* d_out, int out_size, void* d_ws,
                              size_t ws_size, hipStream_t stream) {
  (void)n_in; (void)d_ws; (void)ws_size; (void)out_size;
  const float* x    = (const float*)d_in[0];
  const float* Wq   = (const float*)d_in[1];
  const float* bq   = (const float*)d_in[2];
  const float* Wk   = (const float*)d_in[3];
  const float* bk   = (const float*)d_in[4];
  const float* Wv   = (const float*)d_in[5];
  const float* bv   = (const float*)d_in[6];
  const float* Woa  = (const float*)d_in[7];
  const float* boa  = (const float*)d_in[8];
  const float* Wo   = (const float*)d_in[9];
  const float* bo   = (const float*)d_in[10];
  const float* relk = (const float*)d_in[11];
  const float* relv = (const float*)d_in[12];
  const float* ln1g = (const float*)d_in[13];
  const float* ln1b = (const float*)d_in[14];
  const float* ln2g = (const float*)d_in[15];
  const float* ln2b = (const float*)d_in[16];
  const float* Wf1  = (const float*)d_in[17];
  const float* bf1  = (const float*)d_in[18];
  const float* Wf2  = (const float*)d_in[19];
  const float* bf2  = (const float*)d_in[20];

  const int n_bs = in_sizes[0] / (T_DIM * D_DIM);  // B*S = 1600
  stp_layer_kernel<<<dim3(n_bs), dim3(NTHREADS), 0, stream>>>(
      x, Wq, bq, Wk, bk, Wv, bv, Woa, boa, Wo, bo, relk, relv, ln1g, ln1b,
      ln2g, ln2b, Wf1, bf1, Wf2, bf2, (float*)d_out);
}
